// Attention_67181878444841
// MI455X (gfx1250) — matrix operand order repaired and checked by reading, not compiled, not run on hardware
//
#include <hip/hip_runtime.h>

// ---------------------------------------------------------------------------
// MI455X (gfx1250, wave32) causal multi-head attention forward.
// fp16 WMMA (V_WMMA_F32_16X16X32_F16) everywhere, fp32 accumulation/softmax.
// Pipeline: cvt(x,W) -> GEMM QKV -> flash attention -> GEMM out-proj.
// This revision batches B-fragment LDS loads ahead of WMMA chains so the
// compiler can issue partial s_wait_dscnt instead of a full drain per WMMA.
// ---------------------------------------------------------------------------

typedef __attribute__((ext_vector_type(16))) _Float16 v16h;
typedef __attribute__((ext_vector_type(8)))  _Float16 v8h;
typedef __attribute__((ext_vector_type(8)))  float    v8f;

#define D_MODEL 2048
#define N_HEADS 16
#define D_HEAD  128
#define SEQ     2048
#define BATCH   2
#define MROWS   (BATCH * SEQ)   // 4096

// A-fragment (16x32 f16, M=rows, K=reduction), source row-major, ld in halves.
// lanes 0-15: row m=lane, K = {k0..k0+7, k0+16..k0+23}
// lanes 16-31: row m=lane-16, K = {k0+8..k0+15, k0+24..k0+31}
static __device__ inline v16h load_a_frag(const _Float16* base, int ld,
                                          int row_base, int k0, int lane) {
  const int m   = lane & 15;
  const int sel = (lane >> 4) & 1;
  const _Float16* p = base + (size_t)(row_base + m) * ld + k0 + sel * 8;
  v8h lo = *(const v8h*)p;
  v8h hi = *(const v8h*)(p + 16);
  return __builtin_shufflevector(lo, hi, 0,1,2,3,4,5,6,7,8,9,10,11,12,13,14,15);
}

// B-fragment (32x16 f16, K=reduction, N=cols), source stored TRANSPOSED
// ([n][k] row-major, ld in halves), two 8-half runs per lane (matches A):
// lanes 0-15: col n=lane, K = {k0..k0+7, k0+16..k0+23} ; lanes 16-31: col n=lane-16, K = {k0+8..k0+15, k0+24..k0+31}
static __device__ inline v16h load_b_frag(const _Float16* baseT, int ld,
                                          int n0, int k0, int lane) {
  const int n   = lane & 15;
  const int sel = (lane >> 4) & 1;
  const _Float16* p = baseT + (size_t)(n0 + n) * ld + k0 + sel * 8;
  v8h lo = *(const v8h*)p;
  v8h hi = *(const v8h*)(p + 16);
  return __builtin_shufflevector(lo, hi, 0,1,2,3,4,5,6,7,8,9,10,11,12,13,14,15);
}

// --------------------------- conversion kernels ----------------------------

__global__ void cvt_f2h_kernel(const float* __restrict__ in,
                               _Float16* __restrict__ out, int n) {
  int i = blockIdx.x * blockDim.x + threadIdx.x;
  if (i < n) out[i] = (_Float16)in[i];
}

// W_[QKV]: [h][d][k] fp32 -> WT: [n=h*128+k][d] fp16  (B stored transposed)
__global__ void cvt_wqkvT_kernel(const float* __restrict__ W,
                                 _Float16* __restrict__ WT) {
  int i = blockIdx.x * blockDim.x + threadIdx.x;    // over 2048*2048
  if (i >= D_MODEL * D_MODEL) return;
  int d = i & (D_MODEL - 1);
  int nidx = i >> 11;
  int h = nidx >> 7, k = nidx & 127;
  WT[i] = (_Float16)W[((size_t)h * D_MODEL + d) * D_HEAD + k];
}

// W_O: [h][k][d] fp32 (== [hk][d]) -> WoT: [d][hk] fp16 (transpose)
__global__ void cvt_woT_kernel(const float* __restrict__ W,
                               _Float16* __restrict__ WT) {
  int i = blockIdx.x * blockDim.x + threadIdx.x;
  if (i >= D_MODEL * D_MODEL) return;
  int kk = i & (D_MODEL - 1);
  int d  = i >> 11;
  WT[i] = (_Float16)W[(size_t)kk * D_MODEL + d];
}

// ------------------------------ GEMM kernel --------------------------------
// C[M,N] = A[M,K] @ B[K,N] + bias[N].  A row-major fp16, B given as BT[n][k].
// 256 threads = 8 waves; workgroup tile 128x128; each wave owns a 16x128 slab.
__global__ void __launch_bounds__(256)
gemm_f16(const _Float16* __restrict__ A, const _Float16* __restrict__ BT,
         const float* __restrict__ bias, _Float16* __restrict__ Ch,
         float* __restrict__ Cf, int M, int N, int K, int outHalf) {
  __shared__ _Float16 Ash[128 * 64];
  __shared__ _Float16 Bsh[128 * 64];
  const int tid = threadIdx.x;
  const int lane = tid & 31, wave = tid >> 5;
  const int n0 = blockIdx.x * 128, m0 = blockIdx.y * 128;

  v8f zero = {};
  v8f acc[8];
#pragma unroll
  for (int t = 0; t < 8; ++t) acc[t] = zero;

  for (int kc = 0; kc < K; kc += 64) {
    for (int i = tid; i < 128 * 8; i += 256) {
      int row = i >> 3, ch = i & 7;
      *(v8h*)&Ash[row * 64 + ch * 8] =
          *(const v8h*)&A[(size_t)(m0 + row) * K + kc + ch * 8];
      *(v8h*)&Bsh[row * 64 + ch * 8] =
          *(const v8h*)&BT[(size_t)(n0 + row) * K + kc + ch * 8];
      if (kc + 64 < K) {   // prefetch next panel (global_prefetch_b8)
        __builtin_prefetch(&A[(size_t)(m0 + row) * K + kc + 64 + ch * 8], 0, 3);
        __builtin_prefetch(&BT[(size_t)(n0 + row) * K + kc + 64 + ch * 8], 0, 3);
      }
    }
    __syncthreads();
#pragma unroll
    for (int kk = 0; kk < 64; kk += 32) {
      v16h a = load_a_frag(Ash, 64, wave * 16, kk, lane);
      v16h b[8];
#pragma unroll
      for (int nt = 0; nt < 8; ++nt)
        b[nt] = load_b_frag(Bsh, 64, nt * 16, kk, lane);
#pragma unroll
      for (int nt = 0; nt < 8; ++nt)
        acc[nt] = __builtin_amdgcn_wmma_f32_16x16x32_f16(
            false, a, false, b[nt], (short)0, acc[nt], false, false);
    }
    __syncthreads();
  }

  const int ncol = lane & 15, rhalf = (lane >> 4) & 1;
#pragma unroll
  for (int nt = 0; nt < 8; ++nt) {
    int col = n0 + nt * 16 + ncol;
    float bv = bias ? bias[col] : 0.f;
#pragma unroll
    for (int r = 0; r < 8; ++r) {
      int grow = m0 + wave * 16 + r + rhalf * 8;
      float v = acc[nt][r] + bv;
      if (outHalf) Ch[(size_t)grow * N + col] = (_Float16)v;
      else         Cf[(size_t)grow * N + col] = v;
    }
  }
}

// --------------------------- flash attention -------------------------------
// grid = (S/128 q-blocks, H, B); 256 threads = 8 waves, wave w owns q rows
// [q0+16w, q0+16w+16). Online softmax over key blocks of 64.
__global__ void __launch_bounds__(256)
attn_f16(const _Float16* __restrict__ Qg, const _Float16* __restrict__ Kg,
         const _Float16* __restrict__ Vg, _Float16* __restrict__ Zg) {
  __shared__ _Float16 Ksh[64 * 128];    // [key][d]  (row-major -> B of Q@K^T)
  __shared__ _Float16 VTsh[128 * 64];   // [d][key]  (transposed -> B of P@V)
  __shared__ _Float16 Psh[128 * 64];    // [q][key]  exp(scores), fp16

  const int tid = threadIdx.x, lane = tid & 31, wave = tid >> 5;
  const int q0 = blockIdx.x * 128;
  const int h  = blockIdx.y;
  const int bb = blockIdx.z;
  const int qrb = q0 + wave * 16;
  const float scale = 0.08838834764831845f;   // 1/sqrt(128)

  const _Float16* Qb = Qg + (size_t)bb * SEQ * D_MODEL + h * D_HEAD;
  const _Float16* Kb = Kg + (size_t)bb * SEQ * D_MODEL + h * D_HEAD;
  const _Float16* Vb = Vg + (size_t)bb * SEQ * D_MODEL + h * D_HEAD;

  v16h aQ[4];
#pragma unroll
  for (int kk = 0; kk < 4; ++kk)
    aQ[kk] = load_a_frag(Qb, D_MODEL, qrb, kk * 32, lane);

  float m_r[8], l_r[8];
  v8f zero = {};
  v8f oacc[8];
#pragma unroll
  for (int r = 0; r < 8; ++r) { m_r[r] = -3.0e38f; l_r[r] = 0.f; }
#pragma unroll
  for (int t = 0; t < 8; ++t) oacc[t] = zero;

  const int rhalf = (lane >> 4) & 1, nc = lane & 15;
  const int nkb = q0 / 64 + 2;                // causal: only blocks <= diagonal

  for (int jb = 0; jb < nkb; ++jb) {
    const int j0 = jb * 64;
    // cooperative fill: K row-major, V transposed
    for (int i = tid; i < 64 * 16; i += 256) {
      int row = i >> 4, ch = i & 15;
      v8h kv = *(const v8h*)&Kb[(size_t)(j0 + row) * D_MODEL + ch * 8];
      *(v8h*)&Ksh[row * 128 + ch * 8] = kv;
      v8h vv = *(const v8h*)&Vb[(size_t)(j0 + row) * D_MODEL + ch * 8];
#pragma unroll
      for (int e = 0; e < 8; ++e) VTsh[(ch * 8 + e) * 64 + row] = vv[e];
    }
    __syncthreads();

    // S = Q @ K^T  (16 q x 64 keys per wave, fp32 accum)
    v8f sc[4];
#pragma unroll
    for (int t = 0; t < 4; ++t) sc[t] = zero;
#pragma unroll
    for (int kk = 0; kk < 4; ++kk) {
      v16h bK[4];
#pragma unroll
      for (int t = 0; t < 4; ++t)
        bK[t] = load_b_frag(Ksh, 128, t * 16, kk * 32, lane);
#pragma unroll
      for (int t = 0; t < 4; ++t)
        sc[t] = __builtin_amdgcn_wmma_f32_16x16x32_f16(
            false, aQ[kk], false, bK[t], (short)0, sc[t], false, false);
    }

    // online softmax update (row state replicated across each 16-lane half)
    const bool needmask = (j0 + 63) > qrb;
#pragma unroll
    for (int r = 0; r < 8; ++r) {
      const int qi = qrb + r + rhalf * 8;
      float mx = -3.0e38f;
#pragma unroll
      for (int t = 0; t < 4; ++t) {
        float s = sc[t][r] * scale;
        if (needmask && (j0 + t * 16 + nc) > qi) s = -3.0e38f;
        sc[t][r] = s;
        mx = fmaxf(mx, s);
      }
      mx = fmaxf(mx, __shfl_xor(mx, 1, 32));
      mx = fmaxf(mx, __shfl_xor(mx, 2, 32));
      mx = fmaxf(mx, __shfl_xor(mx, 4, 32));
      mx = fmaxf(mx, __shfl_xor(mx, 8, 32));
      const float mnew = fmaxf(m_r[r], mx);
      const float al = __expf(m_r[r] - mnew);
      m_r[r] = mnew;
      float rs = 0.f;
#pragma unroll
      for (int t = 0; t < 4; ++t) {
        float e = __expf(sc[t][r] - mnew);
        rs += e;
        Psh[(wave * 16 + r + rhalf * 8) * 64 + t * 16 + nc] = (_Float16)e;
      }
      rs += __shfl_xor(rs, 1, 32);
      rs += __shfl_xor(rs, 2, 32);
      rs += __shfl_xor(rs, 4, 32);
      rs += __shfl_xor(rs, 8, 32);
      l_r[r] = l_r[r] * al + rs;
#pragma unroll
      for (int dt = 0; dt < 8; ++dt) oacc[dt][r] *= al;
    }

    // O += P @ V   (P slab is wave-private; Psh RAW handled by DScnt waits)
#pragma unroll
    for (int kk = 0; kk < 2; ++kk) {
      v16h aP = load_a_frag(Psh, 64, wave * 16, kk * 32, lane);
      v16h bV[8];
#pragma unroll
      for (int dt = 0; dt < 8; ++dt)
        bV[dt] = load_b_frag(VTsh, 64, dt * 16, kk * 32, lane);
#pragma unroll
      for (int dt = 0; dt < 8; ++dt)
        oacc[dt] = __builtin_amdgcn_wmma_f32_16x16x32_f16(
            false, aP, false, bV[dt], (short)0, oacc[dt], false, false);
    }
    __syncthreads();   // before next block overwrites Ksh/VTsh
  }

  // finalize: Z = O / l, fp16, layout [b*S + q][h*128 + d]
#pragma unroll
  for (int r = 0; r < 8; ++r) l_r[r] = 1.0f / l_r[r];
#pragma unroll
  for (int dt = 0; dt < 8; ++dt) {
#pragma unroll
    for (int r = 0; r < 8; ++r) {
      int grow = bb * SEQ + qrb + r + rhalf * 8;
      Zg[(size_t)grow * D_MODEL + h * D_HEAD + dt * 16 + nc] =
          (_Float16)(oacc[dt][r] * l_r[r]);
    }
  }
}

// ------------------------------ launcher -----------------------------------

extern "C" void kernel_launch(void* const* d_in, const int* in_sizes, int n_in,
                              void* d_out, int out_size, void* d_ws, size_t ws_size,
                              hipStream_t stream) {
  const float* x   = (const float*)d_in[0];
  const float* W_Q = (const float*)d_in[1];
  const float* b_Q = (const float*)d_in[2];
  const float* W_K = (const float*)d_in[3];
  const float* b_K = (const float*)d_in[4];
  const float* W_V = (const float*)d_in[5];
  const float* b_V = (const float*)d_in[6];
  const float* W_O = (const float*)d_in[7];
  const float* b_O = (const float*)d_in[8];
  float* out = (float*)d_out;

  const size_t MSZ = (size_t)MROWS * D_MODEL;     // 8M halves
  const size_t WSZ = (size_t)D_MODEL * D_MODEL;   // 4M halves
  _Float16* p   = (_Float16*)d_ws;
  _Float16* Xh  = p;            p += MSZ;
  _Float16* WqT = p;            p += WSZ;
  _Float16* WkT = p;            p += WSZ;
  _Float16* WvT = p;            p += WSZ;
  _Float16* WoT = p;            p += WSZ;
  _Float16* Qh  = p;            p += MSZ;
  _Float16* Kh  = p;            p += MSZ;
  _Float16* Vh  = p;            p += MSZ;
  _Float16* Zh  = p;            p += MSZ;

  cvt_f2h_kernel<<<(int)((MSZ + 255) / 256), 256, 0, stream>>>(x, Xh, (int)MSZ);
  cvt_wqkvT_kernel<<<(int)((WSZ + 255) / 256), 256, 0, stream>>>(W_Q, WqT);
  cvt_wqkvT_kernel<<<(int)((WSZ + 255) / 256), 256, 0, stream>>>(W_K, WkT);
  cvt_wqkvT_kernel<<<(int)((WSZ + 255) / 256), 256, 0, stream>>>(W_V, WvT);
  cvt_woT_kernel  <<<(int)((WSZ + 255) / 256), 256, 0, stream>>>(W_O, WoT);

  dim3 gg(D_MODEL / 128, MROWS / 128);   // (16, 32)
  gemm_f16<<<gg, 256, 0, stream>>>(Xh, WqT, b_Q, Qh, nullptr,
                                   MROWS, D_MODEL, D_MODEL, 1);
  gemm_f16<<<gg, 256, 0, stream>>>(Xh, WkT, b_K, Kh, nullptr,
                                   MROWS, D_MODEL, D_MODEL, 1);
  gemm_f16<<<gg, 256, 0, stream>>>(Xh, WvT, b_V, Vh, nullptr,
                                   MROWS, D_MODEL, D_MODEL, 1);

  attn_f16<<<dim3(SEQ / 128, N_HEADS, BATCH), 256, 0, stream>>>(Qh, Kh, Vh, Zh);

  gemm_f16<<<gg, 256, 0, stream>>>(Zh, WoT, b_O, nullptr, out,
                                   MROWS, D_MODEL, D_MODEL, 0);
}
